// LSTM_69303592288677
// MI455X (gfx1250) — matrix operand order repaired and checked by reading, compiled, not run on hardware
//
#include <hip/hip_runtime.h>

typedef unsigned short u16;
typedef __attribute__((ext_vector_type(16))) __bf16 v16bf;
typedef __attribute__((ext_vector_type(8)))  float  v8f;

#define S_LEN 512
#define BATCH 64
#define HID   512
#define GATES 2048   // 4*HID
#define KDIM  512
#define NLAYER 2

// ---------- small device helpers ----------
__device__ __forceinline__ u16 f2bf(float f) {
  union { float f; unsigned u; } x; x.f = f;
  unsigned r = x.u + 0x7FFFu + ((x.u >> 16) & 1u);   // round-to-nearest-even
  return (u16)(r >> 16);
}
__device__ __forceinline__ float sigm(float x) { return 1.0f / (1.0f + __expf(-x)); }

// Low 32 bits of a generic shared-aperture address == LDS byte offset (ISA 10.2).
__device__ __forceinline__ unsigned lds_addr(const void* p) {
  return (unsigned)(unsigned long long)p;
}
// CDNA5 async global->LDS copies (ASYNCcnt-tracked, CDNA5 ISA).
__device__ __forceinline__ void async_b128(unsigned lds, const void* g) {
  asm volatile("global_load_async_to_lds_b128 %0, %1, off" :: "v"(lds), "v"(g) : "memory");
}
__device__ __forceinline__ void async_b32(unsigned lds, const void* g) {
  asm volatile("global_load_async_to_lds_b32 %0, %1, off" :: "v"(lds), "v"(g) : "memory");
}
template <int N>
__device__ __forceinline__ void wait_async() {
  asm volatile("s_wait_asynccnt %0" :: "i"(N) : "memory");
}

// async ops issued per thread per stage (A + B)
template <typename T> struct AsyncOps;
template <> struct AsyncOps<float> { static constexpr int per_stage = 2; }; // B only (A is cvt path)
template <> struct AsyncOps<u16>   { static constexpr int per_stage = 3; }; // A(1) + B(2)

// ---------- weight transpose + f32->bf16 convert ----------
__global__ void transpose_cvt_bf16(const float* __restrict__ w, u16* __restrict__ wT) {
  int n = blockIdx.x;                       // 0..GATES-1
  for (int k = threadIdx.x; k < KDIM; k += blockDim.x)
    wT[(size_t)n * KDIM + k] = f2bf(w[(size_t)k * GATES + n]);
}

// ---------- recurrent state init ----------
__global__ void init_state(const float* __restrict__ h0, const float* __restrict__ c0,
                           float* __restrict__ h, float* __restrict__ c, u16* __restrict__ hbf) {
  int idx = blockIdx.x * blockDim.x + threadIdx.x;   // BATCH*HID
  float hv = h0[idx], cv = c0[idx];
  h[idx] = hv; c[idx] = cv; hbf[idx] = f2bf(hv);
}

// ---------- tile staging ----------
// A tile: MT rows x 32 k, stored with the 8-wide K-block swizzle (blocks 1<->2 swapped)
// so each lane's 16x32 A fragment (ISA 7.12.2) is one contiguous 32B LDS read.
template <int MT>
__device__ __forceinline__ void stage_A_tile(const float* __restrict__ src, u16* ldsA,
                                             size_t rowBase, int k0) {
  constexpr int EPT = (MT * 32) / 256;               // 8 for MT=64
  int linear = threadIdx.x * EPT;
  int r = linear >> 5, kk = linear & 31;
  const float* sp = src + (rowBase + r) * (size_t)KDIM + k0 + kk;
  int blk = kk >> 3, pb = ((blk & 1) << 1) | (blk >> 1);
  u16* dp = &ldsA[r * 32 + (pb << 3) + (kk & 7)];
#pragma unroll
  for (int j = 0; j < EPT; ++j) dp[j] = f2bf(sp[j]);
  if (k0 + 32 < KDIM) __builtin_prefetch(sp + 32, 0, 0);   // global_prefetch_b8
}
template <int MT>
__device__ __forceinline__ void stage_A_tile(const u16* __restrict__ src, u16* ldsA,
                                             size_t rowBase, int k0) {
  constexpr int EPT = (MT * 32) / 256;               // 8 (MT=64) or 2 (MT=16)
  int linear = threadIdx.x * EPT;
  int r = linear >> 5, kk = linear & 31;
  const u16* sp = src + (rowBase + r) * (size_t)KDIM + k0 + kk;
  int blk = kk >> 3, pb = ((blk & 1) << 1) | (blk >> 1);
  unsigned dst = lds_addr(&ldsA[r * 32 + (pb << 3) + (kk & 7)]);
  if constexpr (EPT == 8) async_b128(dst, sp);
  else                    async_b32(dst, sp);
}
// B tile: 128 cols x 32 k, same K-block swizzle as A (blocks 1<->2 swapped; wT is N-major) -> fragment = contiguous 32B.
__device__ __forceinline__ void stage_B_tile(const u16* __restrict__ wT, u16* ldsB,
                                             size_t colBase, int k0) {
  int n  = threadIdx.x >> 1;
  int kk = (threadIdx.x & 1) << 4;
  const u16* gsrc = wT + (colBase + n) * (size_t)KDIM + k0 + kk;
  unsigned dst = lds_addr(&ldsB[n * 32 + (kk >> 1)]);
  async_b128(dst, gsrc);
  async_b128(dst + 32, gsrc + 8);
}

// ---------- WMMA GEMM: dst[M x 2048] = src[M x 512] @ wT^T (+ cbias | crows) ----------
// MT=64: 8 waves as 4(M)x2(N), 4 n-tiles/wave  (big xg GEMM, grid 16 x M/64)
// MT=16: 8 waves as 1(M)x8(N), 1 n-tile /wave  (recurrent step, grid 16 x 4 = 64 WGs)
// Double-buffered LDS; async copy of chunk k+1 overlaps WMMA on chunk k.
template <typename T, int MT>
__global__ __launch_bounds__(256) void gemm_wmma(
    const T*   __restrict__ src,     // M x KDIM
    const u16* __restrict__ wT,      // GATES x KDIM (bf16, N-major)
    const float* __restrict__ cbias, // GATES or nullptr
    const float* __restrict__ crows, // M x GATES or nullptr (C-matrix init)
    float* __restrict__ dst)         // M x GATES
{
  constexpr int NK     = KDIM / 32;   // 16 K-chunks
  constexpr int MWAVES = MT / 16;
  constexpr int NWAVES = 8 / MWAVES;
  constexpr int NTW    = 8 / NWAVES;  // n-tiles per wave
  constexpr int G      = AsyncOps<T>::per_stage;

  __shared__ u16 ldsA[2][MT * 32];
  __shared__ u16 ldsB[2][128 * 32];

  const int lane = threadIdx.x & 31;
  const int wave = threadIdx.x >> 5;
  const int wm   = wave / NWAVES;
  const int wn   = wave % NWAVES;
  const int half = lane >> 4;
  const int l16  = lane & 15;
  const size_t rowBase = (size_t)blockIdx.y * MT;
  const size_t colBase = (size_t)blockIdx.x * 128;

  v8f acc[NTW] = {};
  if (crows) {
#pragma unroll
    for (int nt = 0; nt < NTW; ++nt) {
      size_t n = colBase + wn * (16 * NTW) + nt * 16 + l16;
#pragma unroll
      for (int r = 0; r < 8; ++r) {
        size_t m = rowBase + wm * 16 + half * 8 + r;   // C/D layout: m = 8*half + vgpr
        acc[nt][r] = crows[m * GATES + n];
      }
    }
  }

  // prologue: fill buffer 0
  stage_A_tile<MT>(src, ldsA[0], rowBase, 0);
  stage_B_tile(wT, ldsB[0], colBase, 0);

  for (int kc = 0; kc < NK; ++kc) {
    const int cur = kc & 1;
    if (kc + 1 < NK) {
      // previous release barrier guarantees buf[cur^1] is free
      stage_A_tile<MT>(src, ldsA[cur ^ 1], rowBase, (kc + 1) * 32);
      stage_B_tile(wT, ldsB[cur ^ 1], colBase, (kc + 1) * 32);
      wait_async<G>();      // in-order: <=G outstanding => chunk kc has landed
    } else {
      wait_async<0>();
    }
    __syncthreads();        // data-ready across all waves

    v16bf va = *(const v16bf*)&ldsA[cur][(wm * 16 + l16) * 32 + half * 16];
#pragma unroll
    for (int nt = 0; nt < NTW; ++nt) {
      v16bf vb = *(const v16bf*)&ldsB[cur][(wn * (16 * NTW) + nt * 16 + l16) * 32 + half * 16];
      acc[nt] = __builtin_amdgcn_wmma_f32_16x16x32_bf16(
          false, va, false, vb, (short)0, acc[nt], false, false);
    }
    __syncthreads();        // release buffer
  }

#pragma unroll
  for (int nt = 0; nt < NTW; ++nt) {
    size_t n = colBase + wn * (16 * NTW) + nt * 16 + l16;
    float bv = cbias ? cbias[n] : 0.0f;
#pragma unroll
    for (int r = 0; r < 8; ++r) {
      size_t m = rowBase + wm * 16 + half * 8 + r;
      dst[m * GATES + n] = acc[nt][r] + bv;
    }
  }
}

// ---------- per-step pointwise: 4x LayerNorm + LSTM cell ----------
__global__ __launch_bounds__(512) void lstm_pointwise(
    const float* __restrict__ gates,   // BATCH x GATES
    const float* __restrict__ gamma,   // 4 x HID (this layer)
    const float* __restrict__ beta,    // 4 x HID
    float* __restrict__ h, float* __restrict__ c, u16* __restrict__ hbf,
    u16* __restrict__ ybf,             // nullable: bf16 sequence out (feeds next layer)
    float* __restrict__ yf32,          // nullable: f32 sequence out (final layer)
    float* __restrict__ hs_out,        // nullable: final-h slot (last step)
    float* __restrict__ cs_out)        // nullable: final-c slot (last step)
{
  __shared__ float red[512];
  const int b = blockIdx.x, j = threadIdx.x;

  float g[4], mu[4], rstd[4];
#pragma unroll
  for (int gr = 0; gr < 4; ++gr) g[gr] = gates[(size_t)b * GATES + gr * HID + j];

#pragma unroll
  for (int gr = 0; gr < 4; ++gr) {
    red[j] = g[gr]; __syncthreads();
    for (int s = 256; s > 0; s >>= 1) { if (j < s) red[j] += red[j + s]; __syncthreads(); }
    float sum = red[0]; __syncthreads();
    red[j] = g[gr] * g[gr]; __syncthreads();
    for (int s = 256; s > 0; s >>= 1) { if (j < s) red[j] += red[j + s]; __syncthreads(); }
    float sq = red[0]; __syncthreads();
    float m = sum * (1.0f / HID);
    float var = sq * (1.0f / HID) - m * m;
    mu[gr] = m; rstd[gr] = rsqrtf(var + 1e-5f);
  }

  float nv[4];
#pragma unroll
  for (int gr = 0; gr < 4; ++gr)
    nv[gr] = (g[gr] - mu[gr]) * rstd[gr] * gamma[gr * HID + j] + beta[gr * HID + j];

  const int idx = b * HID + j;
  float cold = c[idx];
  float cn = sigm(nv[1]) * cold + sigm(nv[0]) * tanhf(nv[3]);   // f*c + i*tanh(u)
  float hn = sigm(nv[2]) * tanhf(cn);                           // o*tanh(c)
  c[idx] = cn; h[idx] = hn; hbf[idx] = f2bf(hn);
  if (ybf)  ybf[idx]  = f2bf(hn);
  if (yf32) yf32[idx] = hn;
  if (hs_out) { hs_out[idx] = hn; cs_out[idx] = cn; }
}

// ---------- host orchestration ----------
extern "C" void kernel_launch(void* const* d_in, const int* in_sizes, int n_in,
                              void* d_out, int out_size, void* d_ws, size_t ws_size,
                              hipStream_t stream) {
  (void)in_sizes; (void)n_in; (void)out_size; (void)ws_size;
  const float* inputs   = (const float*)d_in[0];   // S,B,I
  const float* h0       = (const float*)d_in[1];   // L,B,H
  const float* c0       = (const float*)d_in[2];
  const float* wx[2]    = { (const float*)d_in[3], (const float*)d_in[4] };
  const float* wh[2]    = { (const float*)d_in[5], (const float*)d_in[6] };
  const float* bias     = (const float*)d_in[7];   // L,4H
  const float* ln_gamma = (const float*)d_in[8];   // L,4,H
  const float* ln_beta  = (const float*)d_in[9];

  float* out = (float*)d_out;
  float* ys_out = out;                                              // S,B,H
  float* hs_out = out + (size_t)S_LEN * BATCH * HID;                // L,B,H
  float* cs_out = hs_out + (size_t)NLAYER * BATCH * HID;            // L,B,H

  char* wsp = (char*)d_ws;
  auto carve = [&](size_t bytes) -> char* {
    char* p = wsp; wsp += (bytes + 255) & ~(size_t)255; return p;
  };
  u16* wxT[2]; u16* whT[2];
  for (int l = 0; l < 2; ++l) {
    wxT[l] = (u16*)carve((size_t)GATES * KDIM * sizeof(u16));
    whT[l] = (u16*)carve((size_t)GATES * KDIM * sizeof(u16));
  }
  float* xg    = (float*)carve((size_t)S_LEN * BATCH * GATES * sizeof(float)); // 256 MB
  float* gates = (float*)carve((size_t)BATCH * GATES * sizeof(float));
  float* hbuf  = (float*)carve((size_t)BATCH * HID * sizeof(float));
  float* cbuf  = (float*)carve((size_t)BATCH * HID * sizeof(float));
  u16*   hbf   = (u16*)  carve((size_t)BATCH * HID * sizeof(u16));
  u16*   y0bf  = (u16*)  carve((size_t)S_LEN * BATCH * HID * sizeof(u16));     // layer0 seq (bf16)

  // 1) weights -> bf16 transposed
  for (int l = 0; l < 2; ++l) {
    transpose_cvt_bf16<<<GATES, 256, 0, stream>>>(wx[l], wxT[l]);
    transpose_cvt_bf16<<<GATES, 256, 0, stream>>>(wh[l], whT[l]);
  }

  const dim3 gemmGridFull(GATES / 128, (S_LEN * BATCH) / 64);   // 16 x 512
  const dim3 gemmGridStep(GATES / 128, BATCH / 16);             // 16 x 4 = 64 WGs

  for (int l = 0; l < 2; ++l) {
    // 2) xg = X @ wx^T + bias (parallel over all timesteps)
    if (l == 0)
      gemm_wmma<float, 64><<<gemmGridFull, 256, 0, stream>>>(
          inputs, wxT[0], bias + 0 * GATES, nullptr, xg);
    else
      gemm_wmma<u16, 64><<<gemmGridFull, 256, 0, stream>>>(
          y0bf, wxT[1], bias + 1 * GATES, nullptr, xg);

    // 3) state init
    init_state<<<BATCH, HID, 0, stream>>>(h0 + (size_t)l * BATCH * HID,
                                          c0 + (size_t)l * BATCH * HID,
                                          hbuf, cbuf, hbf);

    // 4) recurrent scan (latency-critical: 64 WGs per step via MT=16 tiles)
    const float* gam = ln_gamma + (size_t)l * 4 * HID;
    const float* bet = ln_beta  + (size_t)l * 4 * HID;
    for (int t = 0; t < S_LEN; ++t) {
      gemm_wmma<u16, 16><<<gemmGridStep, 256, 0, stream>>>(
          hbf, whT[l], nullptr, xg + (size_t)t * BATCH * GATES, gates);
      const bool last = (t == S_LEN - 1);
      lstm_pointwise<<<BATCH, HID, 0, stream>>>(
          gates, gam, bet, hbuf, cbuf, hbf,
          (l == 0) ? (y0bf + (size_t)t * BATCH * HID) : nullptr,
          (l == 1) ? (ys_out + (size_t)t * BATCH * HID) : nullptr,
          last ? (hs_out + (size_t)l * BATCH * HID) : nullptr,
          last ? (cs_out + (size_t)l * BATCH * HID) : nullptr);
    }
  }
}
